// CustomLSTM_14740327760565
// MI455X (gfx1250) — matrix operand order repaired and checked by reading, not compiled, not run on hardware
//
#include <hip/hip_runtime.h>
#include <cstdint>
#include <cstddef>

// Problem constants (match reference)
#define B_  64
#define T_  512
#define H_  512
#define L_  4
#define G_  (4 * H_)      // 2048 gate columns
#define BT_ (B_ * T_)     // 32768 rows

typedef __bf16 bf16_t;
typedef __attribute__((ext_vector_type(16))) __bf16 v16bf;
typedef __attribute__((ext_vector_type(8)))  __bf16 v8bf;
typedef __attribute__((ext_vector_type(8)))  float  v8f;

// ---------- helpers ----------

__device__ __forceinline__ bf16_t f2bf(float f) {
  union { float f; unsigned u; } uf; uf.f = f;
  unsigned u = uf.u;
  unsigned r = (u + 0x7FFFu + ((u >> 16) & 1u)) >> 16;   // round-to-nearest-even
  unsigned short s = (unsigned short)r;
  union { unsigned short s; bf16_t b; } ub; ub.s = s;
  return ub.b;
}

__device__ __forceinline__ float fast_rcp(float x) {
#if __has_builtin(__builtin_amdgcn_rcpf)
  return __builtin_amdgcn_rcpf(x);
#else
  return 1.0f / x;
#endif
}

__device__ __forceinline__ float sigmoidf_(float x) {
  return fast_rcp(1.0f + __expf(-x));
}

__device__ __forceinline__ float fast_tanh(float x) {
#if __has_builtin(__builtin_amdgcn_tanhf)
  return __builtin_amdgcn_tanhf(x);    // v_tanh_f32 (CDNA5 hardware transcendental)
#else
  // tanh(x) = 1 - 2/(exp(2x)+1)
  return 1.0f - 2.0f * fast_rcp(__expf(2.0f * x) + 1.0f);
#endif
}

// A fragment (16x32 bf16, M x K) from row-major src with leading dim ld.
// Contiguous-k order (same as load_B): lanes 0-15: M=lane, elems0-15 = K kb..kb+15 (kb=0)
//             lanes 16-31: kb=16.
__device__ __forceinline__ v16bf load_A(const bf16_t* __restrict__ src, int ld, int m0, int k0) {
  int lane = threadIdx.x & 31;
  int m  = m0 + (lane & 15);
  int kb = k0 + ((lane >> 4) << 4);
  const bf16_t* p = src + (size_t)m * ld + kb;
  union { v16bf v; v8bf h[2]; } u;
  u.h[0] = *(const v8bf*)(p);        // K = kb .. kb+7
  u.h[1] = *(const v8bf*)(p + 8);    // K = kb+8 .. kb+15
  return u.v;
}

// B fragment (32x16 bf16, K x N) from row-major N x K weights in global memory.
// Contiguous-k order (same as load_A): lanes 0-15: N=lane, elems0-15 = K k0..k0+15; lanes 16-31: K k0+16..k0+31.
__device__ __forceinline__ v16bf load_B(const bf16_t* __restrict__ w, int ld, int n0, int k0) {
  int lane = threadIdx.x & 31;
  int n  = n0 + (lane & 15);
  int kk = k0 + ((lane >> 4) << 4);
  const bf16_t* p = w + (size_t)n * ld + kk;
  union { v16bf v; v8bf h[2]; } u;
  u.h[0] = *(const v8bf*)(p);
  u.h[1] = *(const v8bf*)(p + 8);
  return u.v;
}

__device__ __forceinline__ v8f wmma_bf16(v16bf a, v16bf b, v8f c) {
  return __builtin_amdgcn_wmma_f32_16x16x32_bf16(false, a, false, b, (short)0, c, false, false);
}

// ---------- conversion / setup kernels ----------

__global__ void k_f32_to_bf16(const float* __restrict__ src, bf16_t* __restrict__ dst, int n) {
  int i = blockIdx.x * blockDim.x + threadIdx.x;
  if (i < n) dst[i] = f2bf(src[i]);
}

__global__ void k_bias_combine(const float* __restrict__ b_ih, const float* __restrict__ b_hh,
                               float* __restrict__ bias, int n) {
  int i = blockIdx.x * blockDim.x + threadIdx.x;
  if (i < n) bias[i] = b_ih[i] + b_hh[i];
}

__global__ void k_init_layer(const float* __restrict__ h0, const float* __restrict__ c0,
                             float* __restrict__ h_state, float* __restrict__ c_state,
                             bf16_t* __restrict__ hbuf, unsigned* __restrict__ bar) {
  int i = blockIdx.x * blockDim.x + threadIdx.x;
  if (i == 0) *bar = 0u;
  if (i < B_ * H_) {
    float h = h0[i], c = c0[i];
    h_state[i] = h;
    c_state[i] = c;
    hbuf[i] = f2bf(h);               // double-buffer slot 0 = initial h
  }
}

// ---------- phase 1: time-parallel input GEMM  pre = in @ W_ih^T  ----------
// in: (B_,T_,H_) bf16 (row r = b*T_+t), w: (G_, H_) bf16 row-major.
// out is stored T-MAJOR: pre[t][b][gate] at ((t*B_+b)*G_ + col) so each recurrent
// step reads one contiguous 512KB slab.
__global__ void k_pregate_gemm(const bf16_t* __restrict__ in, const bf16_t* __restrict__ w,
                               float* __restrict__ out) {
  const int wave = threadIdx.x >> 5;            // 0..7
  const int mt   = blockIdx.y * 8 + wave;       // 0..2047
  const int m0   = mt * 16;                     // row block (fixed b, 16 consecutive t)
  const int n0   = blockIdx.x * 64;             // 4 tiles of 16
  v8f acc0 = {}, acc1 = {}, acc2 = {}, acc3 = {};
  for (int k0 = 0; k0 < H_; k0 += 32) {
    v16bf a  = load_A(in, H_, m0, k0);
    v16bf b0 = load_B(w, H_, n0 +  0, k0);
    v16bf b1 = load_B(w, H_, n0 + 16, k0);
    v16bf b2 = load_B(w, H_, n0 + 32, k0);
    v16bf b3 = load_B(w, H_, n0 + 48, k0);
    acc0 = wmma_bf16(a, b0, acc0);
    acc1 = wmma_bf16(a, b1, acc1);
    acc2 = wmma_bf16(a, b2, acc2);
    acc3 = wmma_bf16(a, b3, acc3);
  }
  const int lane = threadIdx.x & 31;
  const int nn = lane & 15, hi = lane >> 4;
  const int b  = m0 >> 9;           // m0 / T_
  const int t0 = m0 & (T_ - 1);     // m0 % T_  (16-aligned, tile never crosses b)
  v8f accs[4] = {acc0, acc1, acc2, acc3};
#pragma unroll
  for (int j = 0; j < 4; ++j)
#pragma unroll
    for (int r = 0; r < 8; ++r) {
      const int t = t0 + r + 8 * hi;
      out[((size_t)t * B_ + b) * G_ + (n0 + j * 16 + nn)] = accs[j][r];
    }
}

// ---------- phase 2: persistent recurrent kernel (one launch per layer) ----------
// 32 blocks x 128 threads; block nt = blockIdx.x owns hidden cols [16nt,16nt+16),
// wave = mt owns batch rows [16mt,16mt+16). The block's 4-gate W_hh slice (64 KB)
// is staged in LDS once (XOR-swizzled 16B chunks -> conflict-free B-fragment reads),
// then every step reads B only from LDS. c/h state lives in registers all T steps.
#define RNN_BLOCKS  32
#define RNN_THREADS 128

__global__ void k_lstm_recurrent(const float* __restrict__ pre,     // (T_, B_, G_)
                                 const bf16_t* __restrict__ whh,    // (G_, H_)
                                 const float* __restrict__ bias,    // (G_)
                                 const int* __restrict__ lengths,   // (B_)
                                 float* __restrict__ h_state,       // (B_, H_)
                                 float* __restrict__ c_state,       // (B_, H_)
                                 bf16_t* __restrict__ hbuf,         // (2, B_, H_)
                                 bf16_t* __restrict__ act_out,      // (B_, T_, H_)
                                 float* __restrict__ top_out,       // (B_, T_, H_) or null
                                 unsigned* __restrict__ bar) {
  __shared__ __align__(16) bf16_t wlds[4 * 16 * 512];   // 64 KB: [gate][row][K] swizzled

  const int nt   = blockIdx.x;              // 0..31  hidden-col tile
  const int mt   = threadIdx.x >> 5;        // 0..3   batch tile (one wave each)
  const int m0   = mt * 16;
  const int n0   = nt * 16;
  const int lane = threadIdx.x & 31;
  const int nn   = lane & 15, hi = lane >> 4;

  // ---- stage W_hh slice into LDS: 4096 16B chunks, chunk c of row r stored at (c ^ r) ----
  for (int c = threadIdx.x; c < 4096; c += RNN_THREADS) {
    const int rowg  = c >> 6;               // 0..63  (= gate*16 + row)
    const int chunk = c & 63;               // 16B chunk within row
    const int g = rowg >> 4, r = rowg & 15;
    const bf16_t* src = whh + (size_t)(g * H_ + n0 + r) * H_ + chunk * 8;
    char* dst = (char*)wlds + ((size_t)rowg << 10) + ((chunk ^ r) << 4);
    *(v8bf*)dst = *(const v8bf*)src;
  }
  __syncthreads();

  // ---- per-lane persistent state: c/h for (m = m0 + q + 8*hi, col = n0 + nn), q=0..7 ----
  float creg[8], hreg[8];
  int   lenr[8];
  const int col = n0 + nn;
#pragma unroll
  for (int q = 0; q < 8; ++q) {
    const int m = m0 + q + 8 * hi;
    const size_t si = (size_t)m * H_ + col;
    creg[q] = c_state[si];
    hreg[q] = h_state[si];
    lenr[q] = lengths[m];
  }
  float bs[4];
#pragma unroll
  for (int g = 0; g < 4; ++g) bs[g] = bias[g * H_ + col];

  unsigned target = 0;

  for (int t = 0; t < T_; ++t) {
    const bf16_t* hin  = hbuf + (size_t)(t & 1) * (B_ * H_);
    bf16_t*       hout = hbuf + (size_t)((t + 1) & 1) * (B_ * H_);
    const float*  pre_t = pre + (size_t)t * B_ * G_;     // contiguous 512KB slab

    // init accumulators: precomputed input-gates + combined bias
    v8f acc[4];
#pragma unroll
    for (int g = 0; g < 4; ++g) {
#pragma unroll
      for (int q = 0; q < 8; ++q) {
        const int m = m0 + q + 8 * hi;
        acc[g][q] = pre_t[(size_t)m * G_ + g * H_ + col] + bs[g];
      }
    }

    // prefetch next step's pregate cachelines (global_prefetch_b8)
    if (t + 1 < T_) {
      const int mpf = m0 + (lane & 15);
      const float* rowb = pre + (size_t)(t + 1) * B_ * G_ + (size_t)mpf * G_ + n0 + nn;
      __builtin_prefetch(rowb + (hi * 2    ) * H_, 0, 1);
      __builtin_prefetch(rowb + (hi * 2 + 1) * H_, 0, 1);
    }

    // gates += h_{t-1} @ W_hh^T : A from global (L2-hot 64KB), B from LDS
#pragma unroll 4
    for (int k0 = 0; k0 < H_; k0 += 32) {
      v16bf a = load_A(hin, H_, m0, k0);
      const int row = lane & 15;
      const int c0  = (k0 >> 3) + (hi << 1);    // first 16B chunk of this lane's 32B
#pragma unroll
      for (int g = 0; g < 4; ++g) {
        const char* base = (const char*)wlds + (((size_t)(g * 16 + row)) << 10);
        union { v16bf v; v8bf h[2]; } u;
        u.h[0] = *(const v8bf*)(base + (((c0    ) ^ row) << 4));
        u.h[1] = *(const v8bf*)(base + (((c0 + 1) ^ row) << 4));
        acc[g] = wmma_bf16(a, u.v, acc[g]);
      }
    }

    // element-wise LSTM cell + length masking, state kept in registers
#pragma unroll
    for (int q = 0; q < 8; ++q) {
      const int m = m0 + q + 8 * hi;
      const float ig = sigmoidf_(acc[0][q]);
      const float fg = sigmoidf_(acc[1][q]);
      const float gg = fast_tanh(acc[2][q]);
      const float og = sigmoidf_(acc[3][q]);
      float cn = fg * creg[q] + ig * gg;
      float hn = og * fast_tanh(cn);
      const bool live = (t < lenr[q]);
      cn = live ? cn : creg[q];
      hn = live ? hn : hreg[q];
      creg[q] = cn;
      hreg[q] = hn;
      hout[(size_t)m * H_ + col] = f2bf(hn);
      const size_t oi = (size_t)(m * T_ + t) * H_ + col;
      act_out[oi] = f2bf(hn);
      if (top_out) top_out[oi] = hn;
    }

    // one grid-wide barrier per step (protects the h double-buffer swap)
    __threadfence();
    __syncthreads();
    target += RNN_BLOCKS;
    if (threadIdx.x == 0) {
      atomicAdd(bar, 1u);
      while (atomicAdd(bar, 0u) < target) { __builtin_amdgcn_s_sleep(2); }
    }
    __syncthreads();
    __threadfence();
  }

  // write final state back for finalize gather
#pragma unroll
  for (int q = 0; q < 8; ++q) {
    const int m = m0 + q + 8 * hi;
    const size_t si = (size_t)m * H_ + col;
    c_state[si] = creg[q];
    h_state[si] = hreg[q];
  }
}

// ---------- finalize: gather h/c through unsorted_indices ----------
__global__ void k_finalize(const float* __restrict__ h_state, const float* __restrict__ c_state,
                           const int* __restrict__ uidx,
                           float* __restrict__ out_h, float* __restrict__ out_c) {
  int i = blockIdx.x * blockDim.x + threadIdx.x;   // over L_*B_*H_
  if (i >= L_ * B_ * H_) return;
  const int k = i % H_;
  const int b = (i / H_) % B_;
  const int l = i / (H_ * B_);
  const size_t src = ((size_t)l * B_ + uidx[b]) * H_ + k;
  out_h[i] = h_state[src];
  out_c[i] = c_state[src];
}

// ---------- host-side orchestration ----------

static inline size_t align256(size_t x) { return (x + 255) & ~(size_t)255; }

extern "C" void kernel_launch(void* const* d_in, const int* in_sizes, int n_in,
                              void* d_out, int out_size, void* d_ws, size_t ws_size,
                              hipStream_t stream) {
  (void)in_sizes; (void)n_in; (void)out_size;
  const float* x      = (const float*)d_in[0];   // (B,T,H)
  const float* h0     = (const float*)d_in[1];   // (L,B,H)
  const float* c0     = (const float*)d_in[2];   // (L,B,H)
  const float* W_ih   = (const float*)d_in[3];   // (L,4H,H)
  const float* W_hh   = (const float*)d_in[4];   // (L,4H,H)
  const float* b_ih   = (const float*)d_in[5];   // (L,4H)
  const float* b_hh   = (const float*)d_in[6];   // (L,4H)
  const int*   lens   = (const int*)d_in[7];     // (B)
  const int*   uidx   = (const int*)d_in[8];     // (B)

  float* out_seq = (float*)d_out;                              // (B,T,H)
  float* out_h   = out_seq + (size_t)B_ * T_ * H_;             // (L,B,H)
  float* out_c   = out_h   + (size_t)L_ * B_ * H_;             // (L,B,H)

  // workspace carve-up
  char* ws = (char*)d_ws;
  size_t off = 0;
  bf16_t* act0   = (bf16_t*)(ws + off); off = align256(off + (size_t)BT_ * H_ * sizeof(bf16_t));
  bf16_t* act1   = (bf16_t*)(ws + off); off = align256(off + (size_t)BT_ * H_ * sizeof(bf16_t));
  bf16_t* wih_bf = (bf16_t*)(ws + off); off = align256(off + (size_t)L_ * G_ * H_ * sizeof(bf16_t));
  bf16_t* whh_bf = (bf16_t*)(ws + off); off = align256(off + (size_t)L_ * G_ * H_ * sizeof(bf16_t));
  float*  bias   = (float*)(ws + off);  off = align256(off + (size_t)L_ * G_ * sizeof(float));
  float*  pre    = (float*)(ws + off);  off = align256(off + (size_t)BT_ * G_ * sizeof(float));
  float*  hstate = (float*)(ws + off);  off = align256(off + (size_t)L_ * B_ * H_ * sizeof(float));
  float*  cstate = (float*)(ws + off);  off = align256(off + (size_t)L_ * B_ * H_ * sizeof(float));
  bf16_t* hbuf   = (bf16_t*)(ws + off); off = align256(off + (size_t)2 * B_ * H_ * sizeof(bf16_t));
  unsigned* bar  = (unsigned*)(ws + off); off = align256(off + 256);
  if (off > ws_size) return;   // workspace too small: do nothing (deterministic no-op)

  const int CT = 256;
  // conversions (time-parallel, full-chip)
  {
    int n = BT_ * H_;
    k_f32_to_bf16<<<(n + CT - 1) / CT, CT, 0, stream>>>(x, act0, n);
    n = L_ * G_ * H_;
    k_f32_to_bf16<<<(n + CT - 1) / CT, CT, 0, stream>>>(W_ih, wih_bf, n);
    k_f32_to_bf16<<<(n + CT - 1) / CT, CT, 0, stream>>>(W_hh, whh_bf, n);
    n = L_ * G_;
    k_bias_combine<<<(n + CT - 1) / CT, CT, 0, stream>>>(b_ih, b_hh, bias, n);
  }

  for (int l = 0; l < L_; ++l) {
    bf16_t* act_in  = (l & 1) ? act1 : act0;
    bf16_t* act_out = (l & 1) ? act0 : act1;

    // phase 1: pre = act_in @ W_ih[l]^T over all (b,t) rows — fully parallel WMMA GEMM
    dim3 ggrid(G_ / 64, BT_ / (16 * 8));            // (32, 256)
    k_pregate_gemm<<<ggrid, 256, 0, stream>>>(act_in, wih_bf + (size_t)l * G_ * H_, pre);

    // init layer state + barrier counter
    {
      int n = B_ * H_;
      k_init_layer<<<(n + CT - 1) / CT, CT, 0, stream>>>(
          h0 + (size_t)l * B_ * H_, c0 + (size_t)l * B_ * H_,
          hstate + (size_t)l * B_ * H_, cstate + (size_t)l * B_ * H_, hbuf, bar);
    }

    // phase 2: persistent recurrent sweep over T with grid barrier per step
    k_lstm_recurrent<<<RNN_BLOCKS, RNN_THREADS, 0, stream>>>(
        pre, whh_bf + (size_t)l * G_ * H_, bias + (size_t)l * G_, lens,
        hstate + (size_t)l * B_ * H_, cstate + (size_t)l * B_ * H_,
        hbuf, act_out, (l == L_ - 1) ? out_seq : nullptr, bar);
  }

  // gather final states through unsorted_indices
  {
    int n = L_ * B_ * H_;
    k_finalize<<<(n + CT - 1) / CT, CT, 0, stream>>>(hstate, cstate, uidx, out_h, out_c);
  }
}
